// DSC_19207093748017
// MI455X (gfx1250) — matrix operand order repaired and checked by reading, not compiled, not run on hardware
//
#include <hip/hip_runtime.h>

// ---------------------------------------------------------------------------
// MI455X (gfx1250). Live graph (attention branch of the reference is dead):
//   y0  = conv1x1(x, cin_w)                  GEMM M=16384 K=512  N=512
//   d1  = irnn(y0) -> (B,2048,H,W) bf16      4 directional relu scans
//   y1  = conv1x1(d1, c2_w)                  GEMM M=16384 K=2048 N=512
//   d2  = irnn(y1)
//   out = relu(conv1x1(d2, c3_w))            GEMM M=16384 K=2048 N=512
// GEMMs: v_wmma_f32_16x16x32_bf16, 128x128 tile, double-buffered weight tile
// staged with GLOBAL_LOAD_ASYNC_TO_LDS_B128 (ASYNCcnt) when available.
// ---------------------------------------------------------------------------

#define BB 4
#define CC 512
#define HH 64
#define WW 64
#define HW (HH * WW)          // 4096
#define MTOT (BB * HW)        // 16384

#if defined(__has_builtin)
#if __has_builtin(__builtin_amdgcn_global_load_async_to_lds_b128) && \
    __has_builtin(__builtin_amdgcn_s_wait_asynccnt)
#define HAVE_ASYNC 1
#endif
#endif
#ifndef HAVE_ASYNC
#define HAVE_ASYNC 0
#endif

typedef __attribute__((ext_vector_type(8)))  float  v8f;
typedef __attribute__((ext_vector_type(16))) __bf16 v16bf;
typedef __attribute__((ext_vector_type(4)))  int    v4i;
typedef __attribute__((address_space(1))) v4i* gv4i_p;   // global int4*
typedef __attribute__((address_space(3))) v4i* lv4i_p;   // LDS int4*

union Frag16 { uint4 q[2]; v16bf v; unsigned short s[16]; };

static __device__ __forceinline__ unsigned short f2bf(float f) {
    unsigned u = __builtin_bit_cast(unsigned, f);
    u += 0x7FFFu + ((u >> 16) & 1u);          // round-to-nearest-even
    return (unsigned short)(u >> 16);
}
static __device__ __forceinline__ unsigned pack2(float a, float b) {
    return (unsigned)f2bf(a) | ((unsigned)f2bf(b) << 16);
}
static __device__ __forceinline__ float frelu(float v) { return fmaxf(v, 0.0f); }

// ---------------------------------------------------------------------------
// fp32 -> bf16 weight pre-conversion (runs once per launch, 4.7 MB total)
// ---------------------------------------------------------------------------
__global__ __launch_bounds__(256) void cvt_bf16(const float* __restrict__ src,
                                                unsigned short* __restrict__ dst,
                                                int n4) {
    const int i = blockIdx.x * 256 + threadIdx.x;
    if (i < n4) {
        float4 v = *(const float4*)(src + (size_t)i * 4);
        uint2 o;
        o.x = pack2(v.x, v.y);
        o.y = pack2(v.z, v.w);
        *(uint2*)(dst + (size_t)i * 4) = o;
    }
}

// ---------------------------------------------------------------------------
// 1x1 conv as GEMM:  Y[m,n] = sum_k A[m,k] * W[n,k]
//   A: (B, K, HW) activations (fp32 or bf16-bits), W: (512, K) bf16,
//   Y: (B, 512, HW) fp32.
// Workgroup tile 128m x 128n, K-step 32. 8 waves in a 4(m) x 2(n) grid; each
// wave owns 32m x 64n = 2x4 WMMA tiles. Weight tile is double-buffered and
// fetched with async global->LDS b128 copies.
// ---------------------------------------------------------------------------
template <bool AF32, bool RELU>
__global__ __launch_bounds__(256) void gemm1x1(const void* __restrict__ Avoid,
                                               const unsigned short* __restrict__ Wt,
                                               float* __restrict__ Y, int K) {
    __shared__ __align__(16) unsigned char smem[64 * 132 * 4];   // 33792 B
    unsigned short (*Als)[40]  = (unsigned short(*)[40])smem;             // 128x40
    unsigned short (*Wls0)[40] = (unsigned short(*)[40])(smem + 10240);   // 128x40
    unsigned short (*Wls1)[40] = (unsigned short(*)[40])(smem + 20480);   // 128x40
    float (*Ot)[132] = (float(*)[132])smem;                               // 64x132

    const int t    = threadIdx.x;
    const int lane = t & 31;
    const int wv   = t >> 5;
    const int mW   = (wv & 3) * 32;      // wave m origin
    const int nW2  = (wv >> 2) * 64;     // wave n origin (0 or 64)
    const int hlf  = lane >> 4;
    const int l15  = lane & 15;

    const int bIdx   = blockIdx.x >> 5;
    const int hwBase = (blockIdx.x & 31) * 128;
    const int n0     = blockIdx.y * 128;

    const float*          Af = (const float*)Avoid;
    const unsigned short* Ab = (const unsigned short*)Avoid;

    v8f acc[2][4] = {};

    const int amLocal = t & 127;         // A staging: m row
    const int akHalf  = t >> 7;          // A staging: k half (16 ch)
    const int nt      = K >> 5;          // number of 32-wide K tiles

    // W staging chunk map: 128 rows x 4 x 16B chunks = 512 chunks, 2/thread
#define ISSUE_W(k0v, Wb)                                                        \
    {                                                                           \
        _Pragma("unroll")                                                       \
        for (int cch = 0; cch < 2; ++cch) {                                     \
            const int cid = t + cch * 256;                                      \
            const int row = cid >> 2, kq = cid & 3;                             \
            const unsigned short* g = Wt + (size_t)(n0 + row) * K + (k0v) + kq * 8; \
            __builtin_amdgcn_global_load_async_to_lds_b128(                     \
                (gv4i_p)g, (lv4i_p)&(Wb)[row][kq * 8], 0, 0);                   \
        }                                                                       \
    }

#if HAVE_ASYNC
    ISSUE_W(0, Wls0)                     // prologue: tile 0 in flight
#endif

    for (int i = 0; i < nt; ++i) {
        const int k0 = i << 5;
        __syncthreads();                 // previous compute done; LDS reusable

        // ---- stage A tile (128m x 32k) as bf16 into LDS -------------------
        {
            Frag16 pk;
            const size_t rowBase =
                ((size_t)bIdx * K + (size_t)(k0 + akHalf * 16)) * HW + hwBase + amLocal;
            if (AF32) {
#pragma unroll
                for (int kk = 0; kk < 16; ++kk)
                    pk.s[kk] = f2bf(Af[rowBase + (size_t)kk * HW]);
                if (i + 1 < nt) __builtin_prefetch(Af + rowBase + (size_t)32 * HW, 0, 1);
            } else {
#pragma unroll
                for (int kk = 0; kk < 16; ++kk)
                    pk.s[kk] = Ab[rowBase + (size_t)kk * HW];
                if (i + 1 < nt) __builtin_prefetch(Ab + rowBase + (size_t)32 * HW, 0, 1);
            }
            *(uint4*)&Als[amLocal][akHalf * 16]     = pk.q[0];
            *(uint4*)&Als[amLocal][akHalf * 16 + 8] = pk.q[1];
        }

        // ---- weight tile: async double-buffer (or sync fallback) ----------
#if HAVE_ASYNC
        if (i + 1 < nt) {
            if ((i + 1) & 1) { ISSUE_W(k0 + 32, Wls1) } else { ISSUE_W(k0 + 32, Wls0) }
            __builtin_amdgcn_s_wait_asynccnt(2);   // tile i complete, i+1 in flight
        } else {
            __builtin_amdgcn_s_wait_asynccnt(0);
        }
        unsigned short (*Wc)[40] = (i & 1) ? Wls1 : Wls0;
#else
        {
#pragma unroll
            for (int cch = 0; cch < 2; ++cch) {
                const int cid = t + cch * 256;
                const int row = cid >> 2, kq = cid & 3;
                uint4 v = *(const uint4*)(Wt + (size_t)(n0 + row) * K + k0 + kq * 8);
                *(uint4*)&Wls0[row][kq * 8] = v;
            }
        }
        unsigned short (*Wc)[40] = Wls0;
#endif
        __syncthreads();

        // ---- fragments per ISA 7.12.2 layouts -----------------------------
        Frag16 a[2], b[4];
#pragma unroll
        for (int ii = 0; ii < 2; ++ii) {
            const unsigned short* ar = &Als[mW + ii * 16 + l15][0];
            a[ii].q[0] = *(const uint4*)(ar + hlf * 8);
            a[ii].q[1] = *(const uint4*)(ar + 16 + hlf * 8);
        }
#pragma unroll
        for (int j = 0; j < 4; ++j) {
            const unsigned short* br = &Wc[nW2 + j * 16 + l15][0];
            b[j].q[0] = *(const uint4*)(br + hlf * 8);
            b[j].q[1] = *(const uint4*)(br + 16 + hlf * 8);
        }
#pragma unroll
        for (int ii = 0; ii < 2; ++ii)
#pragma unroll
            for (int j = 0; j < 4; ++j)
                acc[ii][j] = __builtin_amdgcn_wmma_f32_16x16x32_bf16(
                    false, a[ii].v, false, b[j].v, (short)0, acc[ii][j], false, false);
    }
#undef ISSUE_W

    // ---- flush: two n-half passes through the LDS transpose buffer --------
#pragma unroll
    for (int p = 0; p < 2; ++p) {
        __syncthreads();
        if ((wv >> 2) == p) {
#pragma unroll
            for (int ii = 0; ii < 2; ++ii)
#pragma unroll
                for (int j = 0; j < 4; ++j) {
                    const int nr = j * 16 + l15;
#pragma unroll
                    for (int r = 0; r < 8; ++r)
                        Ot[nr][mW + ii * 16 + r + hlf * 8] = acc[ii][j][r];
                }
        }
        __syncthreads();
        const int m4 = (t & 31) * 4;
#pragma unroll
        for (int it = 0; it < 8; ++it) {
            const int nl = (t >> 5) + it * 8;
            float4 v = *(const float4*)&Ot[nl][m4];
            if (RELU) { v.x = frelu(v.x); v.y = frelu(v.y); v.z = frelu(v.z); v.w = frelu(v.w); }
            *(float4*)(Y + ((size_t)bIdx * CC + n0 + p * 64 + nl) * HW + hwBase + m4) = v;
        }
    }
}

// ---------------------------------------------------------------------------
// Vertical scans (up & down). Thread owns (b, c, w-pair); lanes along W, so
// loads and packed bf16x2 stores are coalesced. dirs (B,2048,H,W):
// [0,512)=up, [1024,1536)=down.
// ---------------------------------------------------------------------------
__global__ __launch_bounds__(256) void scan_vert(const float* __restrict__ Y,
                                                 const float* __restrict__ wU,
                                                 const float* __restrict__ bU,
                                                 const float* __restrict__ wD,
                                                 const float* __restrict__ bD,
                                                 unsigned short* __restrict__ dirs) {
    const int idx = blockIdx.x * 256 + threadIdx.x;   // B*C*32 threads
    const int w2  = (idx & 31) * 2;
    const int c   = (idx >> 5) & (CC - 1);
    const int b   = idx >> 14;

    const float* xin = Y + ((size_t)b * CC + c) * HW + w2;
    unsigned short* oU = dirs + ((size_t)b * 2048 + c) * HW + w2;
    unsigned short* oD = dirs + ((size_t)b * 2048 + 1024 + c) * HW + w2;

    const float wu = wU[c], bu = bU[c], wd = wD[c], bd = bD[c];

    {   // down: out[0]=0, s=relu(x[0]); s=relu(s*w+b+x[h])
        float2 x0 = *(const float2*)xin;
        float s0 = frelu(x0.x), s1 = frelu(x0.y);
        *(unsigned*)oD = 0u;
        for (int h = 1; h < HH; ++h) {
            float2 xv = *(const float2*)(xin + h * WW);
            s0 = frelu(fmaf(s0, wd, bd) + xv.x);
            s1 = frelu(fmaf(s1, wd, bd) + xv.y);
            *(unsigned*)(oD + h * WW) = pack2(s0, s1);
        }
    }
    {   // up: out[H-1]=0, s=relu(x[H-1]); sweep h descending
        float2 x0 = *(const float2*)(xin + (HH - 1) * WW);
        float s0 = frelu(x0.x), s1 = frelu(x0.y);
        *(unsigned*)(oU + (HH - 1) * WW) = 0u;
        for (int h = HH - 2; h >= 0; --h) {
            float2 xv = *(const float2*)(xin + h * WW);
            s0 = frelu(fmaf(s0, wu, bu) + xv.x);
            s1 = frelu(fmaf(s1, wu, bu) + xv.y);
            *(unsigned*)(oU + h * WW) = pack2(s0, s1);
        }
    }
}

// ---------------------------------------------------------------------------
// Horizontal scans (right & left) via LDS transpose staging: coalesced float4
// in, row scans from LDS (conflict-free padding), packed bf16 stores out.
// [512,1024)=right, [1536,2048)=left.
// ---------------------------------------------------------------------------
__global__ __launch_bounds__(128) void scan_horiz(const float* __restrict__ Y,
                                                  const float* __restrict__ wR,
                                                  const float* __restrict__ bR,
                                                  const float* __restrict__ wL,
                                                  const float* __restrict__ bL,
                                                  unsigned short* __restrict__ dirs) {
    __shared__ float          xin[2][HH][65];
    __shared__ unsigned short rout[2][HH][66];
    __shared__ unsigned short lout[2][HH][66];

    const int t = threadIdx.x;
    const float* src = Y + (size_t)blockIdx.x * (2 * HW);
#pragma unroll
    for (int i = 0; i < 16; ++i) {
        const int e = (i * 128 + t) * 4;
        float4 v = *(const float4*)(src + e);
        const int p = e >> 12, rem = e & 4095, h = rem >> 6, w = rem & 63;
        xin[p][h][w] = v.x; xin[p][h][w + 1] = v.y;
        xin[p][h][w + 2] = v.z; xin[p][h][w + 3] = v.w;
    }
    __syncthreads();

    const int p = t >> 6, h = t & 63;
    const int plane = blockIdx.x * 2 + p;
    const int c = plane & (CC - 1);
    const float wr = wR[c], br = bR[c], wl = wL[c], bl = bL[c];

    {   // right
        float s = frelu(xin[p][h][0]);
        rout[p][h][0] = 0;
        for (int w = 1; w < WW; ++w) {
            s = frelu(fmaf(s, wr, br) + xin[p][h][w]);
            rout[p][h][w] = f2bf(s);
        }
    }
    {   // left
        float s = frelu(xin[p][h][WW - 1]);
        lout[p][h][WW - 1] = 0;
        for (int w = WW - 2; w >= 0; --w) {
            s = frelu(fmaf(s, wl, bl) + xin[p][h][w]);
            lout[p][h][w] = f2bf(s);
        }
    }
    __syncthreads();

#pragma unroll
    for (int i = 0; i < 32; ++i) {
        const int pe = i * 128 + t;
        const int pp = pe >> 11, rem = pe & 2047, hh = rem >> 5, j = rem & 31;
        const int pl = blockIdx.x * 2 + pp;
        const int cc = pl & (CC - 1), bb = pl >> 9;
        const unsigned rv = *(const unsigned*)&rout[pp][hh][2 * j];
        const unsigned lv = *(const unsigned*)&lout[pp][hh][2 * j];
        const size_t eb = (size_t)bb * 2048 * HW + (size_t)hh * WW + 2 * j;
        *(unsigned*)(dirs + eb + (size_t)(512 + cc) * HW)  = rv;
        *(unsigned*)(dirs + eb + (size_t)(1536 + cc) * HW) = lv;
    }
}

// ---------------------------------------------------------------------------
extern "C" void kernel_launch(void* const* d_in, const int* in_sizes, int n_in,
                              void* d_out, int out_size, void* d_ws, size_t ws_size,
                              hipStream_t stream) {
    (void)in_sizes; (void)n_in; (void)out_size; (void)ws_size;
    const float* x     = (const float*)d_in[0];
    const float* cin_w = (const float*)d_in[7];
    const float* c2_w  = (const float*)d_in[8];
    const float* c3_w  = (const float*)d_in[9];
    const float* i1_wu = (const float*)d_in[10]; const float* i1_bu = (const float*)d_in[11];
    const float* i1_wr = (const float*)d_in[12]; const float* i1_br = (const float*)d_in[13];
    const float* i1_wd = (const float*)d_in[14]; const float* i1_bd = (const float*)d_in[15];
    const float* i1_wl = (const float*)d_in[16]; const float* i1_bl = (const float*)d_in[17];
    const float* i2_wu = (const float*)d_in[18]; const float* i2_bu = (const float*)d_in[19];
    const float* i2_wr = (const float*)d_in[20]; const float* i2_br = (const float*)d_in[21];
    const float* i2_wd = (const float*)d_in[22]; const float* i2_bd = (const float*)d_in[23];
    const float* i2_wl = (const float*)d_in[24]; const float* i2_bl = (const float*)d_in[25];

    // workspace: y fp32 (33.5 MB) | dirs bf16 (67 MB) | bf16 weights (4.7 MB)
    float*          y    = (float*)d_ws;
    unsigned short* dirs = (unsigned short*)((char*)d_ws + (size_t)BB * CC * HW * 4);
    unsigned short* wbf  = dirs + (size_t)BB * 4 * CC * HW;
    unsigned short* wb_cin = wbf;
    unsigned short* wb_c2  = wb_cin + (size_t)CC * CC;
    unsigned short* wb_c3  = wb_c2 + (size_t)CC * 4 * CC;

    cvt_bf16<<<(CC * CC / 4 + 255) / 256, 256, 0, stream>>>(cin_w, wb_cin, CC * CC / 4);
    cvt_bf16<<<(CC * 4 * CC / 4 + 255) / 256, 256, 0, stream>>>(c2_w, wb_c2, CC * CC);
    cvt_bf16<<<(CC * 4 * CC / 4 + 255) / 256, 256, 0, stream>>>(c3_w, wb_c3, CC * CC);

    const dim3 gg(MTOT / 128, CC / 128);   // 128 x 4 workgroups

    gemm1x1<true,  false><<<gg, 256, 0, stream>>>(x, wb_cin, y, CC);
    scan_vert <<<BB * CC * (WW / 2) / 256, 256, 0, stream>>>(y, i1_wu, i1_bu, i1_wd, i1_bd, dirs);
    scan_horiz<<<BB * CC / 2, 128, 0, stream>>>(y, i1_wr, i1_br, i1_wl, i1_bl, dirs);

    gemm1x1<false, false><<<gg, 256, 0, stream>>>(dirs, wb_c2, y, 4 * CC);
    scan_vert <<<BB * CC * (WW / 2) / 256, 256, 0, stream>>>(y, i2_wu, i2_bu, i2_wd, i2_bd, dirs);
    scan_horiz<<<BB * CC / 2, 128, 0, stream>>>(y, i2_wr, i2_br, i2_wl, i2_bl, dirs);

    gemm1x1<false, true ><<<gg, 256, 0, stream>>>(dirs, wb_c3, (float*)d_out, 4 * CC);
}
